// GNN_13125420056773
// MI455X (gfx1250) — compile-verified, matrix operand order checked by reading, not run on hardware
//
#include <hip/hip_runtime.h>
#include <hip/hip_bf16.h>

typedef __attribute__((ext_vector_type(16))) _Float16 v16h;
typedef __attribute__((ext_vector_type(8)))  _Float16 v8h;
typedef __attribute__((ext_vector_type(8)))  float    v8f;

#define NNODE  16384
#define NEDGE  262144
#define CEMB   300
#define CHID   600
#define NLAYER 5
#define KP1    320      // CEMB padded to mult of 32
#define KP2    608      // CHID padded to mult of 32
#define NP1    608      // CHID padded to mult of 16/32 (reused as K of gemm2)
#define NP2    304      // CEMB padded to mult of 16
#define MBLK   4        // M-tiles per wave (B-fragment reuse factor)

// ---------------------------------------------------------------------------
// Zero-fill
// ---------------------------------------------------------------------------
__global__ void gnn_zero_f32(float* __restrict__ p, int n) {
    int i = blockIdx.x * blockDim.x + threadIdx.x;
    if (i < n) p[i] = 0.0f;
}

// ---------------------------------------------------------------------------
// Initial node embedding: sum of six table lookups -> h [NNODE, CEMB]
// ---------------------------------------------------------------------------
__global__ void gnn_init_embed(const int* __restrict__ x,
                               const float* __restrict__ ea,
                               const float* __restrict__ ed,
                               const float* __restrict__ ec,
                               const float* __restrict__ eh,
                               const float* __restrict__ ear,
                               const float* __restrict__ ech,
                               float* __restrict__ h) {
    int n = blockIdx.x;
    int i0 = x[n * 6 + 0];
    int i1 = x[n * 6 + 1];
    int i2 = x[n * 6 + 2];
    int i3 = x[n * 6 + 3];
    int i4 = x[n * 6 + 4];
    int i5 = x[n * 6 + 5];
    const float* r0 = ea  + (size_t)i0 * CEMB;
    const float* r1 = ed  + (size_t)i1 * CEMB;
    const float* r2 = ec  + (size_t)i2 * CEMB;
    const float* r3 = eh  + (size_t)i3 * CEMB;
    const float* r4 = ear + (size_t)i4 * CEMB;
    const float* r5 = ech + (size_t)i5 * CEMB;
    float* out = h + (size_t)n * CEMB;
    for (int c = threadIdx.x; c < CEMB; c += blockDim.x)
        out[c] = r0[c] + r1[c] + r2[c] + r3[c] + r4[c] + r5[c];
}

// ---------------------------------------------------------------------------
// Message + scatter-add: one block per edge (incl. self loops).
// agg[dst] += h[src] + edge_emb1[a0] + edge_emb2[a1]   (float atomics -> L2)
// ---------------------------------------------------------------------------
__global__ void gnn_scatter(const float* __restrict__ h,
                            const int* __restrict__ edge_index,
                            const int* __restrict__ edge_attr,
                            const float* __restrict__ ee1,   // [6, CEMB]
                            const float* __restrict__ ee2,   // [3, CEMB]
                            float* __restrict__ agg) {
    int e = blockIdx.x;
    int src, dst, a0, a1;
    if (e < NEDGE) {
        src = edge_index[e];
        dst = edge_index[NEDGE + e];
        a0  = edge_attr[2 * e + 0];
        a1  = edge_attr[2 * e + 1];
    } else {                      // self loop: attr = (4, 0)
        src = dst = e - NEDGE;
        a0 = 4; a1 = 0;
    }
    const float* hs = h   + (size_t)src * CEMB;
    const float* t1 = ee1 + (size_t)a0  * CEMB;
    const float* t2 = ee2 + (size_t)a1  * CEMB;
    float* ag       = agg + (size_t)dst * CEMB;
    for (int c = threadIdx.x; c < CEMB; c += blockDim.x)
        atomicAdd(&ag[c], hs[c] + t1[c] + t2[c]);
}

// ---------------------------------------------------------------------------
// Pack f32 activation [M, C] -> padded f16 [M, CP]  (pad columns = 0)
// ---------------------------------------------------------------------------
__global__ void gnn_pack_a_f16(const float* __restrict__ src,
                               _Float16* __restrict__ dst, int M, int C, int CP) {
    int i = blockIdx.x * blockDim.x + threadIdx.x;
    if (i >= M * CP) return;
    int m = i / CP, c = i - m * CP;
    dst[i] = (c < C) ? (_Float16)src[(size_t)m * C + c] : (_Float16)0.0f;
}

// ---------------------------------------------------------------------------
// Pack + transpose f32 weight [K, Nv] -> padded f16 B^T [NP, KP]
// dst[n*KP + k] = src[k*Nv + n]
// ---------------------------------------------------------------------------
__global__ void gnn_pack_bt_f16(const float* __restrict__ src,
                                _Float16* __restrict__ dst,
                                int K, int Nv, int KPv, int NPv) {
    int i = blockIdx.x * blockDim.x + threadIdx.x;
    if (i >= NPv * KPv) return;
    int n = i / KPv, k = i - n * KPv;
    dst[i] = (n < Nv && k < K) ? (_Float16)src[(size_t)k * Nv + n] : (_Float16)0.0f;
}

// ---------------------------------------------------------------------------
// WMMA GEMM, register-blocked: each wave computes MBLK 16x16 tiles in M
// sharing one B fragment (4x B reuse).  K tiled by 32, fully branchless.
// A: f16 [M, KPAD] row-major (pad zeroed).  BT: f16 [NPAD, KPAD] (B^T).
// F16OUT: write f16 [M, NPAD] (pads zeroed, chains as next A).
// else:   write f32 [M, Nvalid] (guarded).
// Fragment layout per CDNA5 ISA 7.12.2: each lane's 16 halves are two
// contiguous groups of 8 in K -> two global_load_b128 per operand per chunk.
// ---------------------------------------------------------------------------
template <int KPAD, int NPAD, bool F16OUT>
__global__ void gnn_wmma_gemm(const _Float16* __restrict__ A,
                              const _Float16* __restrict__ BT,
                              const float* __restrict__ bias,
                              _Float16* __restrict__ Ch,
                              float* __restrict__ Cf,
                              int M, int Nvalid, int relu) {
    const int lane = threadIdx.x & 31;
    const int wave = threadIdx.x >> 5;
    const int w    = blockIdx.x * (blockDim.x >> 5) + wave;
    const int tilesN  = NPAD >> 4;
    const int mblocks = M / (16 * MBLK);
    if (w >= mblocks * tilesN) return;         // wave-uniform: EXEC stays all-1s
    const int tmb = w / tilesN;                // M super-tile (MBLK * 16 rows)
    const int tn  = w - tmb * tilesN;

    const int half16 = lane >> 4;              // 0: lanes 0-15, 1: lanes 16-31
    const int l15    = lane & 15;
    const int kbase  = half16 * 8;
    const int ncol   = tn * 16 + l15;          // B column this lane supplies
    const int mrow0  = tmb * (16 * MBLK) + l15;

    const _Float16* bcol  = BT + (size_t)ncol * KPAD + kbase;
    const _Float16* arow0 = A  + (size_t)mrow0 * KPAD + kbase;

    v8f acc[MBLK];
#pragma unroll
    for (int j = 0; j < MBLK; ++j) acc[j] = (v8f){};

#pragma unroll 2
    for (int kc = 0; kc < KPAD; kc += 32) {
        v8h blo = *(const v8h*)(bcol + kc);        // K = kc+kbase .. +7
        v8h bhi = *(const v8h*)(bcol + kc + 16);   // K = kc+kbase+16 .. +23
        v16h b = __builtin_shufflevector(blo, bhi,
                 0,1,2,3,4,5,6,7,8,9,10,11,12,13,14,15);
#pragma unroll
        for (int j = 0; j < MBLK; ++j) {
            const _Float16* ar = arow0 + (size_t)j * 16 * KPAD;
            v8h alo = *(const v8h*)(ar + kc);
            v8h ahi = *(const v8h*)(ar + kc + 16);
            v16h a = __builtin_shufflevector(alo, ahi,
                     0,1,2,3,4,5,6,7,8,9,10,11,12,13,14,15);
            acc[j] = __builtin_amdgcn_wmma_f32_16x16x32_f16(
                /*neg_a=*/false, a, /*neg_b=*/false, b,
                /*c_mod=*/(short)0, acc[j], /*reuse_a=*/false, /*reuse_b=*/false);
        }
    }

    if constexpr (F16OUT) {
        const bool nok = (ncol < Nvalid);
        float bv = nok ? bias[ncol] : 0.0f;
#pragma unroll
        for (int j = 0; j < MBLK; ++j) {
#pragma unroll
            for (int r = 0; r < 8; ++r) {
                int m = tmb * (16 * MBLK) + j * 16 + r + half16 * 8;  // C/D layout
                float v = acc[j][r] + bv;
                if (relu) v = fmaxf(v, 0.0f);
                Ch[(size_t)m * NPAD + ncol] = nok ? (_Float16)v : (_Float16)0.0f;
            }
        }
    } else {
        if (ncol < Nvalid) {
            float bv = bias[ncol];
#pragma unroll
            for (int j = 0; j < MBLK; ++j) {
#pragma unroll
                for (int r = 0; r < 8; ++r) {
                    int m = tmb * (16 * MBLK) + j * 16 + r + half16 * 8;
                    float v = acc[j][r] + bv;
                    if (relu) v = fmaxf(v, 0.0f);
                    Cf[(size_t)m * Nvalid + ncol] = v;
                }
            }
        }
    }
}

// ---------------------------------------------------------------------------
// BatchNorm1d over nodes (training stats, biased var) + affine + optional ReLU
// ---------------------------------------------------------------------------
__global__ void gnn_bn_relu(float* __restrict__ h,
                            const float* __restrict__ gamma,
                            const float* __restrict__ beta,
                            int N, int Cc, int relu) {
    const int c = blockIdx.x;
    __shared__ float s_sum[256];
    __shared__ float s_sq[256];
    float sum = 0.0f, sq = 0.0f;
    for (int n = threadIdx.x; n < N; n += blockDim.x) {
        float v = h[(size_t)n * Cc + c];
        sum += v;
        sq  += v * v;
    }
    s_sum[threadIdx.x] = sum;
    s_sq[threadIdx.x]  = sq;
    __syncthreads();
    for (int s = blockDim.x >> 1; s > 0; s >>= 1) {
        if ((int)threadIdx.x < s) {
            s_sum[threadIdx.x] += s_sum[threadIdx.x + s];
            s_sq[threadIdx.x]  += s_sq[threadIdx.x + s];
        }
        __syncthreads();
    }
    const float invN = 1.0f / (float)N;
    const float mu   = s_sum[0] * invN;
    const float var  = s_sq[0] * invN - mu * mu;
    const float scl  = rsqrtf(var + 1e-5f) * gamma[c];
    const float sh   = beta[c] - mu * scl;
    for (int n = threadIdx.x; n < N; n += blockDim.x) {
        float v = h[(size_t)n * Cc + c] * scl + sh;
        if (relu) v = fmaxf(v, 0.0f);
        h[(size_t)n * Cc + c] = v;
    }
}

// ---------------------------------------------------------------------------
// Host orchestration
// ---------------------------------------------------------------------------
extern "C" void kernel_launch(void* const* d_in, const int* in_sizes, int n_in,
                              void* d_out, int out_size, void* d_ws, size_t ws_size,
                              hipStream_t stream) {
    const int*   x          = (const int*)  d_in[0];
    const int*   edge_index = (const int*)  d_in[1];
    const int*   edge_attr  = (const int*)  d_in[2];
    const float* emb_atom   = (const float*)d_in[3];
    const float* emb_degree = (const float*)d_in[4];
    const float* emb_charge = (const float*)d_in[5];
    const float* emb_hybrid = (const float*)d_in[6];
    const float* emb_arom   = (const float*)d_in[7];
    const float* emb_chir   = (const float*)d_in[8];
    const float* W1         = (const float*)d_in[9];   // [L, 300, 600]
    const float* b1         = (const float*)d_in[10];  // [L, 600]
    const float* W2         = (const float*)d_in[11];  // [L, 600, 300]
    const float* b2         = (const float*)d_in[12];  // [L, 300]
    const float* ee1        = (const float*)d_in[13];  // [L, 6, 300]
    const float* ee2        = (const float*)d_in[14];  // [L, 3, 300]
    const float* bn_gamma   = (const float*)d_in[15];  // [L, 300]
    const float* bn_beta    = (const float*)d_in[16];  // [L, 300]

    float* h = (float*)d_out;                          // [NNODE, CEMB]

    // workspace carve-up (all regions 256-byte aligned by construction)
    char* ws = (char*)d_ws;
    float*    agg  = (float*)ws;     ws += (size_t)NNODE * CEMB * sizeof(float);
    _Float16* aggh = (_Float16*)ws;  ws += (size_t)NNODE * KP1 * sizeof(_Float16);
    _Float16* midh = (_Float16*)ws;  ws += (size_t)NNODE * NP1 * sizeof(_Float16);
    _Float16* w1t  = (_Float16*)ws;  ws += (size_t)NLAYER * NP1 * KP1 * sizeof(_Float16);
    _Float16* w2t  = (_Float16*)ws;  ws += (size_t)NLAYER * NP2 * KP2 * sizeof(_Float16);

    // pre-pack all layers' weights: W^T, f16, padded
    {
        int n1 = NP1 * KP1;  // per-layer elements of w1t
        int n2 = NP2 * KP2;  // per-layer elements of w2t
        for (int l = 0; l < NLAYER; ++l) {
            gnn_pack_bt_f16<<<(n1 + 255) / 256, 256, 0, stream>>>(
                W1 + (size_t)l * CEMB * CHID, w1t + (size_t)l * n1, CEMB, CHID, KP1, NP1);
            gnn_pack_bt_f16<<<(n2 + 255) / 256, 256, 0, stream>>>(
                W2 + (size_t)l * CHID * CEMB, w2t + (size_t)l * n2, CHID, CEMB, KP2, NP2);
        }
    }

    // initial embedding
    gnn_init_embed<<<NNODE, 128, 0, stream>>>(x, emb_atom, emb_degree, emb_charge,
                                              emb_hybrid, emb_arom, emb_chir, h);

    const int aggElems  = NNODE * CEMB;
    const int agghElems = NNODE * KP1;
    const int tiles1 = (NNODE / (16 * MBLK)) * (NP1 / 16);   // 256 * 38
    const int tiles2 = (NNODE / (16 * MBLK)) * (NP2 / 16);   // 256 * 19
    const int blocks1 = (tiles1 + 7) / 8;                    // 8 waves per block
    const int blocks2 = (tiles2 + 7) / 8;

    for (int l = 0; l < NLAYER; ++l) {
        gnn_zero_f32<<<(aggElems + 255) / 256, 256, 0, stream>>>(agg, aggElems);

        gnn_scatter<<<NEDGE + NNODE, 128, 0, stream>>>(
            h, edge_index, edge_attr,
            ee1 + (size_t)l * 6 * CEMB,
            ee2 + (size_t)l * 3 * CEMB,
            agg);

        gnn_pack_a_f16<<<(agghElems + 255) / 256, 256, 0, stream>>>(
            agg, aggh, NNODE, CEMB, KP1);

        // midh = relu(agg @ W1[l] + b1[l])  (f16 out, padded)
        gnn_wmma_gemm<KP1, NP1, true><<<blocks1, 256, 0, stream>>>(
            aggh, w1t + (size_t)l * NP1 * KP1, b1 + (size_t)l * CHID,
            midh, (float*)nullptr, NNODE, CHID, /*relu=*/1);

        // h = mid @ W2[l] + b2[l]  (f32 out)
        gnn_wmma_gemm<KP2, NP2, false><<<blocks2, 256, 0, stream>>>(
            midh, w2t + (size_t)l * NP2 * KP2, b2 + (size_t)l * CEMB,
            (_Float16*)nullptr, h, NNODE, CEMB, /*relu=*/0);

        // batchnorm (+relu except last layer), in place on h
        gnn_bn_relu<<<CEMB, 256, 0, stream>>>(
            h, bn_gamma + (size_t)l * CEMB, bn_beta + (size_t)l * CEMB,
            NNODE, CEMB, l < NLAYER - 1 ? 1 : 0);
    }
}
